// DiffusionDecoder_8340826488878
// MI455X (gfx1250) — matrix operand order repaired and checked by reading, not compiled, not run on hardware
//
#include <hip/hip_runtime.h>
#include <hip/hip_bf16.h>

typedef _Float16 half_t;
typedef __attribute__((ext_vector_type(16))) _Float16 v16h;
typedef __attribute__((ext_vector_type(8)))  _Float16 v8h;
typedef __attribute__((ext_vector_type(8)))  float    v8f;

#define HDIM 128
#define WAVES_PER_BLOCK 4

// ---------------------------------------------------------------------------
// Exact float atomic-max via integer atomics (IEEE total order trick):
//   value >= 0  -> signed   atomicMax on bit pattern
//   value <  0  -> unsigned atomicMin on bit pattern
// Destination initialized to -inf (0xFF800000).
// ---------------------------------------------------------------------------
__device__ __forceinline__ void atomic_max_f32(float* addr, float v) {
  if (v >= 0.0f) atomicMax((int*)addr, __float_as_int(v));
  else           atomicMin((unsigned int*)addr, __float_as_uint(v));
}

// ---------------------------------------------------------------------------
// Pack W1 [L,256,128] f32 -> WMMA-B fragment-contiguous f16 tiles.
// Fragment (kk,nt): 32 lanes x 16 halves contiguous. Lane ln carries
// column n = nt*16 + (ln&15), K = kk*32 + (ln>>4)*8 + (h&7) + (h>>3)*16  (h = 0..15).
// ---------------------------------------------------------------------------
__global__ void pack_w1_kernel(const float* __restrict__ W1,
                               half_t* __restrict__ w1t, int total) {
  int tid = blockIdx.x * blockDim.x + threadIdx.x;
  if (tid >= total) return;
  int idx = tid;
  int h  = idx & 15; idx >>= 4;
  int ln = idx & 31; idx >>= 5;
  int nt = idx & 7;  idx >>= 3;
  int kk = idx & 7;  idx >>= 3;
  int l  = idx;
  int k = kk * 32 + (ln >> 4) * 8 + (h & 7) + (h >> 3) * 16;   // 0..255
  int n = nt * 16 + (ln & 15);            // 0..127
  w1t[tid] = (half_t)W1[((size_t)l * 256 + k) * HDIM + n];
}

__global__ void pack_w2_kernel(const float* __restrict__ W2,
                               half_t* __restrict__ w2t, int total) {
  int tid = blockIdx.x * blockDim.x + threadIdx.x;
  if (tid >= total) return;
  int idx = tid;
  int h  = idx & 15; idx >>= 4;
  int ln = idx & 31; idx >>= 5;
  int nt = idx & 7;  idx >>= 3;
  int kk = idx & 3;  idx >>= 2;
  int l  = idx;
  int k = kk * 32 + (ln >> 4) * 8 + (h & 7) + (h >> 3) * 16;   // 0..127
  int n = nt * 16 + (ln & 15);            // 0..127
  w2t[tid] = (half_t)W2[((size_t)l * 128 + k) * HDIM + n];
}

__global__ void f32_to_f16_kernel(const float* __restrict__ in,
                                  half_t* __restrict__ out, int total) {
  int i = blockIdx.x * blockDim.x + threadIdx.x;
  if (i < total) out[i] = (half_t)in[i];
}

__global__ void fill_neg_inf_kernel(float* __restrict__ p, int total) {
  int i = blockIdx.x * blockDim.x + threadIdx.x;
  if (i < total) p[i] = -__builtin_huge_valf();
}

// ---------------------------------------------------------------------------
// EdgeConv layer: one wave32 per 16-edge tile.
//   msg[16x256] = [x_i | x_j - x_i]  (f16, LDS)
//   h = relu(msg @ W1 + b1)          (WMMA f16->f32, 64 wmma)
//   y = h @ W2 + b2                  (WMMA f16->f32, 32 wmma)
//   out[dst] = max(out[dst], y)      (atomic f32 max)
// ---------------------------------------------------------------------------
__global__ __launch_bounds__(WAVES_PER_BLOCK * 32)
void edge_mlp_max_kernel(const half_t* __restrict__ xh,
                         const int*    __restrict__ ei,   // [2,E]
                         const half_t* __restrict__ w1t,  // layer-tiled
                         const float*  __restrict__ b1,
                         const half_t* __restrict__ w2t,
                         const float*  __restrict__ b2,
                         float* __restrict__ out,
                         int nTiles, int nEdges, int nTotal) {
  __shared__ half_t lds_msg[WAVES_PER_BLOCK][16 * 256];
  __shared__ half_t lds_a2 [WAVES_PER_BLOCK][16 * 128];
  __shared__ int    lds_dst[WAVES_PER_BLOCK][16];

  const int wave = threadIdx.x >> 5;
  const int lane = threadIdx.x & 31;
  int tile = blockIdx.x * WAVES_PER_BLOCK + wave;
  const bool valid = (tile < nTiles);
  if (!valid) tile = nTiles - 1;   // clamp so all waves reach barriers

  // ---- destination indices for the 16 tile rows ----
  if (lane < 16) {
    int e = tile * 16 + lane;
    int d = -1;
    if (e < nEdges)      d = ei[nEdges + e];
    else if (e < nTotal) d = e - nEdges;     // self loop
    lds_dst[wave][lane] = d;
  }

  // ---- gather node rows, build msg = [x_i | x_j - x_i] in LDS ----
  {
    const int r  = lane >> 1;          // tile row 0..15 (2 lanes per row)
    const int c0 = (lane & 1) * 64;    // column half
    int e = tile * 16 + r;
    int sN = 0, dN = 0;
    if (e < nEdges)      { sN = ei[e]; dN = ei[nEdges + e]; }
    else if (e < nTotal) { sN = dN = e - nEdges; }
    const v8h* pi = (const v8h*)(xh + (size_t)dN * HDIM + c0);  // x_i = x[dst]
    const v8h* pj = (const v8h*)(xh + (size_t)sN * HDIM + c0);  // x_j = x[src]
    half_t* mrow = &lds_msg[wave][r * 256];
#pragma unroll
    for (int t = 0; t < 8; ++t) {
      v8h a = pi[t];
      v8h b = pj[t];
      *(v8h*)(mrow + c0 + t * 8)       = a;       // x_i
      *(v8h*)(mrow + 128 + c0 + t * 8) = b - a;   // x_j - x_i
    }
  }
  __syncthreads();

  const int kgrp = lane >> 4;   // A-fragment K-half select
  const int nIdx = lane & 15;   // C/D column within n-tile
  const int m    = lane & 15;   // A-fragment row

  const v8f vzero = {0.f, 0.f, 0.f, 0.f, 0.f, 0.f, 0.f, 0.f};

  // ---- GEMM1: 16x256 @ 256x128 ----
  v8f acc[8];
#pragma unroll
  for (int nt = 0; nt < 8; ++nt) acc[nt] = vzero;
#pragma unroll
  for (int kk = 0; kk < 8; ++kk) {
    const int koff = kk * 32 + kgrp * 8;
    v8h alo = *(const v8h*)(&lds_msg[wave][m * 256 + koff]);        // K koff..+7
    v8h ahi = *(const v8h*)(&lds_msg[wave][m * 256 + koff + 16]);   // K koff+16..+23
    v16h a = __builtin_shufflevector(alo, ahi, 0,1,2,3,4,5,6,7,8,9,10,11,12,13,14,15);
#pragma unroll
    for (int nt = 0; nt < 8; ++nt) {
      v16h b = *(const v16h*)(w1t + ((size_t)(kk * 8 + nt) * 32 + lane) * 16);
      acc[nt] = __builtin_amdgcn_wmma_f32_16x16x32_f16(
          false, a, false, b, (short)0, acc[nt], false, false);
    }
  }

  // ---- bias + ReLU -> LDS (f16) as GEMM2 A-matrix ----
#pragma unroll
  for (int nt = 0; nt < 8; ++nt) {
    float bias = b1[nt * 16 + nIdx];
#pragma unroll
    for (int v = 0; v < 8; ++v) {
      float val = acc[nt][v] + bias;
      val = fmaxf(val, 0.0f);
      lds_a2[wave][(v + 8 * kgrp) * HDIM + nt * 16 + nIdx] = (half_t)val;
    }
  }
  __syncthreads();

  // ---- GEMM2: 16x128 @ 128x128 ----
  v8f acc2[8];
#pragma unroll
  for (int nt = 0; nt < 8; ++nt) acc2[nt] = vzero;
#pragma unroll
  for (int kk = 0; kk < 4; ++kk) {
    const int koff = kk * 32 + kgrp * 8;
    v8h alo = *(const v8h*)(&lds_a2[wave][m * HDIM + koff]);
    v8h ahi = *(const v8h*)(&lds_a2[wave][m * HDIM + koff + 16]);
    v16h a = __builtin_shufflevector(alo, ahi, 0,1,2,3,4,5,6,7,8,9,10,11,12,13,14,15);
#pragma unroll
    for (int nt = 0; nt < 8; ++nt) {
      v16h b = *(const v16h*)(w2t + ((size_t)(kk * 8 + nt) * 32 + lane) * 16);
      acc2[nt] = __builtin_amdgcn_wmma_f32_16x16x32_f16(
          false, a, false, b, (short)0, acc2[nt], false, false);
    }
  }

  // ---- bias + atomic segment-max scatter ----
  if (valid) {
#pragma unroll
    for (int nt = 0; nt < 8; ++nt) {
      float bias = b2[nt * 16 + nIdx];
#pragma unroll
      for (int v = 0; v < 8; ++v) {
        int node = lds_dst[wave][v + 8 * kgrp];
        if (node >= 0)
          atomic_max_f32(out + (size_t)node * HDIM + nt * 16 + nIdx,
                         acc2[nt][v] + bias);
      }
    }
  }
}

// ---------------------------------------------------------------------------
// Final projection: out[N,3] = x[N,128] @ Wf[128,3] + bf
// ---------------------------------------------------------------------------
__global__ void final_fc_kernel(const float* __restrict__ x,
                                const float* __restrict__ Wf,
                                const float* __restrict__ bf,
                                float* __restrict__ out, int N) {
  int i = blockIdx.x * blockDim.x + threadIdx.x;
  if (i >= N) return;
  const float* row = x + (size_t)i * HDIM;
  float s0 = bf[0], s1 = bf[1], s2 = bf[2];
#pragma unroll 4
  for (int k = 0; k < HDIM; ++k) {
    float xv = row[k];
    s0 += xv * Wf[k * 3 + 0];
    s1 += xv * Wf[k * 3 + 1];
    s2 += xv * Wf[k * 3 + 2];
  }
  out[i * 3 + 0] = s0;
  out[i * 3 + 1] = s1;
  out[i * 3 + 2] = s2;
}

// ---------------------------------------------------------------------------
extern "C" void kernel_launch(void* const* d_in, const int* in_sizes, int n_in,
                              void* d_out, int out_size, void* d_ws, size_t ws_size,
                              hipStream_t stream) {
  const float* x  = (const float*)d_in[0];
  const int*   ei = (const int*)d_in[1];
  const float* W1 = (const float*)d_in[2];
  const float* b1 = (const float*)d_in[3];
  const float* W2 = (const float*)d_in[4];
  const float* b2 = (const float*)d_in[5];
  const float* Wf = (const float*)d_in[6];
  const float* bf = (const float*)d_in[7];

  const int N  = in_sizes[0] / HDIM;     // 50000
  const int E  = in_sizes[1] / 2;        // 800000
  const int L  = in_sizes[3] / HDIM;     // 4 (b1 is [L,H])
  const int ET = E + N;                  // edges + self loops
  const int nTiles = (ET + 15) / 16;

  const size_t W1T_PER_LAYER = (size_t)8 * 8 * 32 * 16;   // halves
  const size_t W2T_PER_LAYER = (size_t)4 * 8 * 32 * 16;   // halves

  char* ws = (char*)d_ws;
  size_t off = 0;
  half_t* w1t = (half_t*)(ws + off); off += L * W1T_PER_LAYER * sizeof(half_t);
  half_t* w2t = (half_t*)(ws + off); off += L * W2T_PER_LAYER * sizeof(half_t);
  half_t* xh  = (half_t*)(ws + off); off += (size_t)N * HDIM * sizeof(half_t);
  float*  bufA = (float*)(ws + off); off += (size_t)N * HDIM * sizeof(float);
  float*  bufB = (float*)(ws + off); off += (size_t)N * HDIM * sizeof(float);
  (void)ws_size;

  // Repack weights every call (deterministic; ~0.4 MB total).
  {
    int total1 = (int)(L * W1T_PER_LAYER);
    pack_w1_kernel<<<(total1 + 255) / 256, 256, 0, stream>>>(W1, w1t, total1);
    int total2 = (int)(L * W2T_PER_LAYER);
    pack_w2_kernel<<<(total2 + 255) / 256, 256, 0, stream>>>(W2, w2t, total2);
  }

  const int totalNH = N * HDIM;
  const float* cur = x;
  for (int l = 0; l < L; ++l) {
    float* nxt = (l & 1) ? bufB : bufA;
    f32_to_f16_kernel<<<(totalNH + 255) / 256, 256, 0, stream>>>(cur, xh, totalNH);
    fill_neg_inf_kernel<<<(totalNH + 255) / 256, 256, 0, stream>>>(nxt, totalNH);
    int blocks = (nTiles + WAVES_PER_BLOCK - 1) / WAVES_PER_BLOCK;
    edge_mlp_max_kernel<<<blocks, WAVES_PER_BLOCK * 32, 0, stream>>>(
        xh, ei,
        w1t + (size_t)l * W1T_PER_LAYER, b1 + (size_t)l * HDIM,
        w2t + (size_t)l * W2T_PER_LAYER, b2 + (size_t)l * HDIM,
        nxt, nTiles, E, ET);
    cur = nxt;
  }

  final_fc_kernel<<<(N + 127) / 128, 128, 0, stream>>>(cur, Wf, bf, (float*)d_out, N);
}
